// RNN_28656021799190
// MI455X (gfx1250) — matrix operand order repaired and checked by reading, not compiled, not run on hardware
//
#include <hip/hip_runtime.h>

// ---------------------------------------------------------------------------
// RNN: xp = x @ W_xh + b ; h_t = tanh(xp_t + h_{t-1} @ W_hh)
// B=64, T=2048, IN=H=512.  bf16 WMMA (v_wmma_f32_16x16x32_bf16), f32 accum.
// Phase 1: tiled WMMA GEMM writes xp into d_out.
// Phase 2: 4 independent blocks (16 batch rows each) run the whole scan with
//          h resident in LDS (bf16) and W_hh streamed from L2 in
//          fragment-major layout (coalesced b128 loads, zero shuffles).
// ---------------------------------------------------------------------------

#define BB 64
#define TT 2048
#define KIN 512
#define HH 512

typedef __attribute__((ext_vector_type(16))) __bf16 v16bf;
typedef __attribute__((ext_vector_type(8)))  float  v8f;

union FragU {
    v16bf v;
    uint4 q[2];
    unsigned int   u32[8];
    unsigned short u16[16];
};

__device__ __forceinline__ unsigned short f2bf(float x) {
    unsigned int u = __float_as_uint(x);
    // round-to-nearest-even truncation to bf16
    u += 0x7FFFu + ((u >> 16) & 1u);
    return (unsigned short)(u >> 16);
}

__device__ __forceinline__ v8f wmma_bf16(v16bf a, v16bf b, v8f c) {
    return __builtin_amdgcn_wmma_f32_16x16x32_bf16(
        /*neg_a=*/false, a, /*neg_b=*/false, b,
        /*c_mod=*/(short)0, c, /*reuse_a=*/false, /*reuse_b=*/false);
}

// ---------------------------------------------------------------------------
// Kernel 1: convert W_xh / W_hh (fp32 row-major [k][n]) into fragment-major
// bf16: fragment f = nt*16 + kc holds the exact 32-lane x 16-bf16 WMMA
// B-operand image, contiguous (1 KB per fragment, lane-stride 32 bytes).
// B layout (16-bit, 32x16): lane = half*16 + n%16; element j (v=j>>1,p=j&1):
//   k = kc*32 + (j<8 ? half*8 + j : 16 + half*8 + j-8) ; col n = nt*16 + (lane&15)
// 1024 waves total (2 matrices x 32 ntiles x 16 kchunks).
// ---------------------------------------------------------------------------
__global__ void swizzle_w_kernel(const float* __restrict__ W_xh,
                                 const float* __restrict__ W_hh,
                                 unsigned short* __restrict__ w_sw) {
    int gwave = (blockIdx.x * blockDim.x + threadIdx.x) >> 5;
    int lane  = threadIdx.x & 31;
    int mat   = gwave >> 9;           // 0 = W_xh, 1 = W_hh
    int f     = gwave & 511;          // nt*16 + kc
    int kc    = f & 15;
    int hf    = lane >> 4;
    int n     = ((f >> 4) << 4) + (lane & 15);
    const float* W = mat ? W_hh : W_xh;
    unsigned short* dst = w_sw + (size_t)mat * (512 * 512) + (size_t)f * 512 + lane * 16;
#pragma unroll
    for (int j = 0; j < 16; ++j) {
        int v = j >> 1, p = j & 1;
        int k = kc * 32 + ((v < 4) ? (hf * 8 + 2 * v + p) : (16 + hf * 8 + 2 * (v - 4) + p));
        dst[j] = f2bf(W[(size_t)k * HH + n]);
    }
}

// ---------------------------------------------------------------------------
// Kernel 2: xp = x @ W_xh + b  ->  d_out   (131072 x 512)
// Block: 256 threads (8 waves = 4 M-subtiles x 2 N-subtiles), tile 64x64.
// A (x) staged fp32->bf16 through LDS (coalesced b128 global loads);
// B fragments loaded directly from the swizzled weights (2x b128 per frag).
// ---------------------------------------------------------------------------
__global__ void xp_gemm_kernel(const float* __restrict__ x,
                               const unsigned short* __restrict__ wxh_sw,
                               const float* __restrict__ bias,
                               float* __restrict__ out) {
    __shared__ unsigned short As[64][40];   // 64 rows x 32 k, pitch 40 (pad)

    const int tid  = threadIdx.x;
    const int lane = tid & 31;
    const int wave = tid >> 5;
    const int l15  = lane & 15;
    const int hf   = lane >> 4;
    const int wm   = wave >> 1;             // 0..3  (M sub-tile)
    const int wn   = wave & 1;              // 0..1  (N sub-tile, 32 cols)
    const int row0 = blockIdx.x * 64;       // over B*T
    const int n0b  = blockIdx.y * 64;       // over H

    v8f c0 = {}, c1 = {};

    const int srow = tid >> 2;              // staging: 0..63
    const int scol = (tid & 3) * 8;         // 0,8,16,24

    for (int kc = 0; kc < 16; ++kc) {
        __syncthreads();
        {   // stage 64x32 fp32 tile of x as bf16 into LDS
            const float* src = x + (size_t)(row0 + srow) * KIN + kc * 32 + scol;
            float4 a = *(const float4*)src;
            float4 b = *(const float4*)(src + 4);
            unsigned short* d = &As[srow][scol];
            d[0] = f2bf(a.x); d[1] = f2bf(a.y); d[2] = f2bf(a.z); d[3] = f2bf(a.w);
            d[4] = f2bf(b.x); d[5] = f2bf(b.y); d[6] = f2bf(b.z); d[7] = f2bf(b.w);
        }
        __syncthreads();

        // A fragment (16-bit A 16x32 layout), packed-pair ds_load_b32 gathers
        FragU a;
        const int m = wm * 16 + l15;
#pragma unroll
        for (int v = 0; v < 8; ++v) {
            int k = (v < 4) ? (hf * 8 + 2 * v) : (16 + hf * 8 + 2 * (v - 4));
            a.u32[v] = *(const unsigned int*)&As[m][k];
        }

        // B fragments: two 16-col tiles, fully coalesced b128 pairs
        const int ntb = (n0b >> 4) + wn * 2;
        const unsigned short* f0 = wxh_sw + ((size_t)(ntb + 0) * 16 + kc) * 512 + lane * 16;
        const unsigned short* f1 = wxh_sw + ((size_t)(ntb + 1) * 16 + kc) * 512 + lane * 16;
        FragU b0, b1;
        b0.q[0] = *(const uint4*)f0; b0.q[1] = *(const uint4*)(f0 + 8);
        b1.q[0] = *(const uint4*)f1; b1.q[1] = *(const uint4*)(f1 + 8);

        c0 = wmma_bf16(a.v, b0.v, c0);
        c1 = wmma_bf16(a.v, b1.v, c1);
    }

    // epilogue: + bias, store fp32 (D layout: row r+8*half, col lane%16)
    const int n_0 = n0b + wn * 32 + l15;
    const int n_1 = n_0 + 16;
    const float bb0 = bias[n_0];
    const float bb1 = bias[n_1];
#pragma unroll
    for (int r = 0; r < 8; ++r) {
        const size_t row = (size_t)(row0 + wm * 16 + r + 8 * hf);
        out[row * HH + n_0] = c0[r] + bb0;
        out[row * HH + n_1] = c1[r] + bb1;
    }
}

// ---------------------------------------------------------------------------
// Kernel 3: sequential scan. 4 blocks x 256 threads (8 waves).
// Block owns batch rows [b0, b0+16); h kept in LDS as bf16 (pitch 520).
// Each wave owns 64 H columns (4 N-tiles); per step: 4x16 = 64 WMMAs.
// out[b][t][:] is read (xp) and overwritten (h_t) in place.
// ---------------------------------------------------------------------------
__global__ void rnn_scan_kernel(const float* __restrict__ h0,
                                const unsigned short* __restrict__ whh_sw,
                                float* __restrict__ out) {
    __shared__ unsigned short Hs[16][520];  // 16 x 512 bf16, padded pitch

    const int tid  = threadIdx.x;
    const int lane = tid & 31;
    const int wave = tid >> 5;
    const int l15  = lane & 15;
    const int hf   = lane >> 4;
    const int b0   = blockIdx.x * 16;
    const int nt0  = wave * 4;              // first of 4 owned N-tiles

    // load initial hidden state
    for (int i = tid; i < 16 * 512; i += 256) {
        int r = i >> 9, ccol = i & 511;
        Hs[r][ccol] = f2bf(h0[(size_t)(b0 + r) * HH + ccol]);
    }
    __syncthreads();

    float hv[4][8];

    for (int t = 0; t < TT; ++t) {
        v8f c0 = {}, c1 = {}, c2 = {}, c3 = {};

        for (int kc = 0; kc < 16; ++kc) {
            FragU a;
#pragma unroll
            for (int v = 0; v < 8; ++v) {
                int k = kc * 32 + ((v < 4) ? (hf * 8 + 2 * v)
                                           : (16 + hf * 8 + 2 * (v - 4)));
                a.u32[v] = *(const unsigned int*)&Hs[l15][k];
            }
            const unsigned short* fb =
                whh_sw + ((size_t)nt0 * 16 + kc) * 512 + lane * 16;
            FragU b0f, b1f, b2f, b3f;
            b0f.q[0] = *(const uint4*)(fb);             b0f.q[1] = *(const uint4*)(fb + 8);
            b1f.q[0] = *(const uint4*)(fb + 16 * 512);  b1f.q[1] = *(const uint4*)(fb + 16 * 512 + 8);
            b2f.q[0] = *(const uint4*)(fb + 32 * 512);  b2f.q[1] = *(const uint4*)(fb + 32 * 512 + 8);
            b3f.q[0] = *(const uint4*)(fb + 48 * 512);  b3f.q[1] = *(const uint4*)(fb + 48 * 512 + 8);
            c0 = wmma_bf16(a.v, b0f.v, c0);
            c1 = wmma_bf16(a.v, b1f.v, c1);
            c2 = wmma_bf16(a.v, b2f.v, c2);
            c3 = wmma_bf16(a.v, b3f.v, c3);
        }

        // h_t = tanh(xp + acc); overwrite out in place
#pragma unroll
        for (int i = 0; i < 4; ++i) {
            const int n = (nt0 + i) * 16 + l15;
#pragma unroll
            for (int r = 0; r < 8; ++r) {
                const int b = r + 8 * hf;
                const size_t idx = ((size_t)(b0 + b) * TT + t) * HH + n;
                float acc = (i == 0 ? c0[r] : i == 1 ? c1[r] : i == 2 ? c2[r] : c3[r]);
                float v = tanhf(out[idx] + acc);
                out[idx] = v;
                hv[i][r] = v;
            }
        }

        __syncthreads();   // all waves done reading Hs for this step
#pragma unroll
        for (int i = 0; i < 4; ++i) {
            const int n = (nt0 + i) * 16 + l15;
#pragma unroll
            for (int r = 0; r < 8; ++r)
                Hs[r + 8 * hf][n] = f2bf(hv[i][r]);
        }
        __syncthreads();   // Hs updated before next step's reads
    }

    // final hidden state h_n, fp32, appended after outputs
    float* hn = out + (size_t)BB * TT * HH;
#pragma unroll
    for (int i = 0; i < 4; ++i) {
        const int n = (nt0 + i) * 16 + l15;
#pragma unroll
        for (int r = 0; r < 8; ++r)
            hn[(size_t)(b0 + r + 8 * hf) * HH + n] = hv[i][r];
    }
}

// ---------------------------------------------------------------------------
extern "C" void kernel_launch(void* const* d_in, const int* in_sizes, int n_in,
                              void* d_out, int out_size, void* d_ws, size_t ws_size,
                              hipStream_t stream) {
    const float* x    = (const float*)d_in[0];
    const float* h0   = (const float*)d_in[1];
    const float* W_xh = (const float*)d_in[2];
    const float* b_xh = (const float*)d_in[3];
    const float* W_hh = (const float*)d_in[4];
    float* out = (float*)d_out;

    unsigned short* w_sw = (unsigned short*)d_ws;      // [0,512K): W_xh, [512K,1M): W_hh
    (void)in_sizes; (void)n_in; (void)out_size; (void)ws_size;

    // 1) fragment-major bf16 weight swizzle (1024 waves)
    swizzle_w_kernel<<<128, 256, 0, stream>>>(W_xh, W_hh, w_sw);

    // 2) xp = x @ W_xh + b -> d_out
    dim3 g2(2048, 8);
    xp_gemm_kernel<<<g2, 256, 0, stream>>>(x, w_sw, b_xh, out);

    // 3) sequential scan, 4 independent batch-tiles
    rnn_scan_kernel<<<4, 256, 0, stream>>>(h0, w_sw + 512 * 512, out);
}
